// model_1_1013612282299
// MI455X (gfx1250) — matrix operand order repaired and checked by reading, not compiled, not run on hardware
//
#include <hip/hip_runtime.h>
#include <math.h>

typedef __attribute__((ext_vector_type(16))) _Float16 v16h;
typedef __attribute__((ext_vector_type(8)))  _Float16 v8h;
typedef __attribute__((ext_vector_type(8)))  float    v8f;

// ---------------------------------------------------------------------------
// 1. Embedding gather: x[n][d] = emb[x_idx[n]][d]
// ---------------------------------------------------------------------------
__global__ void gin_embed(const int* __restrict__ x_idx,
                          const float* __restrict__ emb,
                          float* __restrict__ X, int total) {
  int i = blockIdx.x * blockDim.x + threadIdx.x;
  if (i >= total) return;
  int n = i >> 7;
  int d = i & 127;
  X[i] = emb[(long)x_idx[n] * 128 + d];
}

// ---------------------------------------------------------------------------
// 2. Pack a 128x128 fp32 weight matrix into WMMA B-operand order (f16).
//    For tile (kt,nt), lane l needs 16 halves: K = kt*32 + (l<16?0:8) + (j&7) + (j>=8?16:0),
//    N = nt*16 + (l&15).  Stored contiguously (32B per lane) for one load.
// ---------------------------------------------------------------------------
__global__ void gin_pack_w(const float* __restrict__ W, _Float16* __restrict__ out) {
  int t = blockIdx.x * blockDim.x + threadIdx.x;   // 16384 threads
  if (t >= 4 * 8 * 32 * 16) return;
  int j    = t & 15;
  int lane = (t >> 4) & 31;
  int nt   = (t >> 9) & 7;
  int kt   = t >> 12;
  int K = kt * 32 + ((lane < 16) ? 0 : 8) + (j & 7) + ((j & 8) << 1);
  int n = nt * 16 + (lane & 15);
  out[t] = (_Float16)W[K * 128 + n];
}

// ---------------------------------------------------------------------------
// 3. Init aggregation buffer: h = (2 + eps) * x   (folds self-loop w=1 and
//    the (1+eps)*x term of the GIN update)
// ---------------------------------------------------------------------------
__global__ void gin_init_h(float* __restrict__ H, const float* __restrict__ X,
                           const float* __restrict__ eps, int total) {
  int i = blockIdx.x * blockDim.x + threadIdx.x;
  if (i >= total) return;
  H[i] = (2.0f + eps[0]) * X[i];
}

// ---------------------------------------------------------------------------
// 4. Edge scatter: h[dst] += w_e * x[src]   (one thread per (edge, channel))
//    All traffic stays in the 192MB L2 (node buffer is 25.6MB).
// ---------------------------------------------------------------------------
__global__ void gin_scatter(float* __restrict__ H, const float* __restrict__ X,
                            const int* __restrict__ src, const int* __restrict__ dst,
                            const float* __restrict__ w, long total) {
  long i = (long)blockIdx.x * blockDim.x + threadIdx.x;
  if (i >= total) return;
  int e = (int)(i >> 7);
  int d = (int)(i & 127);
  int s = src[e];
  int t = dst[e];
  atomicAdd(&H[(long)t * 128 + d], w[e] * X[(long)s * 128 + d]);
}

// ---------------------------------------------------------------------------
// 5. Fused 2-layer MLP with WMMA:
//    X_out = relu( relu(H @ W1 + b1) @ W2 + b2 )
//    8 waves / block, each wave owns one 16x16 N-tile; h1 staged in LDS (f16).
// ---------------------------------------------------------------------------
__global__ __launch_bounds__(256) void gin_mlp(
    const float* __restrict__ H,
    const _Float16* __restrict__ Wp1, const float* __restrict__ b1,
    const _Float16* __restrict__ Wp2, const float* __restrict__ b2,
    float* __restrict__ Xo, int N) {
  __shared__ _Float16 lds_h1[16 * 128];

  const int lane = threadIdx.x & 31;
  const int nt   = threadIdx.x >> 5;          // 0..7 : N-tile of this wave
  const int M0   = blockIdx.x * 16;
  const int mrow = lane & 15;
  const int koff = (lane < 16) ? 0 : 8;       // 16-bit A-layout K split
  const int arow = M0 + mrow;
  const int arow_c = (arow < N) ? arow : (N - 1);   // clamp loads (no EXEC div.)

  // ---- GEMM1: h1 = relu(H @ W1 + b1), one 16x16 tile per wave ----
  v8f c = {};
#pragma unroll
  for (int kt = 0; kt < 4; ++kt) {
    const float* pa = H + (long)arow_c * 128 + kt * 32 + koff;
    float4 f0 = *(const float4*)(pa);
    float4 f1 = *(const float4*)(pa + 4);
    float4 f2 = *(const float4*)(pa + 16);
    float4 f3 = *(const float4*)(pa + 20);
    v16h a;
    a[0] = (_Float16)f0.x;  a[1] = (_Float16)f0.y;
    a[2] = (_Float16)f0.z;  a[3] = (_Float16)f0.w;
    a[4] = (_Float16)f1.x;  a[5] = (_Float16)f1.y;
    a[6] = (_Float16)f1.z;  a[7] = (_Float16)f1.w;
    a[8]  = (_Float16)f2.x; a[9]  = (_Float16)f2.y;
    a[10] = (_Float16)f2.z; a[11] = (_Float16)f2.w;
    a[12] = (_Float16)f3.x; a[13] = (_Float16)f3.y;
    a[14] = (_Float16)f3.z; a[15] = (_Float16)f3.w;
    v16h b = *(const v16h*)(Wp1 + (((kt * 8 + nt) * 32 + lane) << 4));
    c = __builtin_amdgcn_wmma_f32_16x16x32_f16(false, a, false, b,
                                               (short)0, c, false, false);
  }

  // bias + relu, deposit h1 tile into LDS (row-major f16 [16][128])
  const float bias1 = b1[nt * 16 + mrow];
#pragma unroll
  for (int r = 0; r < 8; ++r) {
    float v = c[r] + bias1;
    v = v > 0.0f ? v : 0.0f;
    int row = r + ((lane < 16) ? 0 : 8);
    lds_h1[row * 128 + nt * 16 + mrow] = (_Float16)v;
  }
  __syncthreads();

  // ---- GEMM2: out = relu(h1 @ W2 + b2) ----
  v8f c2 = {};
#pragma unroll
  for (int kt = 0; kt < 4; ++kt) {
    v8h lo = *(const v8h*)&lds_h1[mrow * 128 + kt * 32 + koff];
    v8h hi = *(const v8h*)&lds_h1[mrow * 128 + kt * 32 + 16 + koff];
    v16h a = __builtin_shufflevector(lo, hi, 0, 1, 2, 3, 4, 5, 6, 7,
                                     8, 9, 10, 11, 12, 13, 14, 15);
    v16h b = *(const v16h*)(Wp2 + (((kt * 8 + nt) * 32 + lane) << 4));
    c2 = __builtin_amdgcn_wmma_f32_16x16x32_f16(false, a, false, b,
                                                (short)0, c2, false, false);
  }

  const float bias2 = b2[nt * 16 + mrow];
#pragma unroll
  for (int r = 0; r < 8; ++r) {
    float v = c2[r] + bias2;
    v = v > 0.0f ? v : 0.0f;
    int row = M0 + r + ((lane < 16) ? 0 : 8);
    if (row < N) Xo[(long)row * 128 + nt * 16 + mrow] = v;
  }
}

// ---------------------------------------------------------------------------
// 6. Graph pooling
// ---------------------------------------------------------------------------
__global__ void gin_zero_pool(float* __restrict__ gsum, float* __restrict__ cnt, int G) {
  int i = blockIdx.x * blockDim.x + threadIdx.x;
  if (i < G * 128) gsum[i] = 0.0f;
  if (i < G) cnt[i] = 0.0f;
}

__global__ void gin_pool(const float* __restrict__ X, const int* __restrict__ batch,
                         float* __restrict__ gsum, float* __restrict__ cnt, int total) {
  int i = blockIdx.x * blockDim.x + threadIdx.x;
  if (i >= total) return;
  int n = i >> 7;
  int d = i & 127;
  int b = batch[n];
  atomicAdd(&gsum[b * 128 + d], X[i]);
  if (d == 0) atomicAdd(&cnt[b], 1.0f);
}

// ---------------------------------------------------------------------------
// 7. Head: out = softmax( relu(g @ fW1 + fb1) @ fW2 + fb2 ), g = gsum/cnt
//    (tiny: 64x128x64 + 64x64x2 — scalar per graph row)
// ---------------------------------------------------------------------------
__global__ void gin_head(const float* __restrict__ gsum, const float* __restrict__ cnt,
                         const float* __restrict__ fW1, const float* __restrict__ fb1,
                         const float* __restrict__ fW2, const float* __restrict__ fb2,
                         float* __restrict__ out, int G) {
  int t = blockIdx.x * blockDim.x + threadIdx.x;
  if (t >= G) return;
  float c = cnt[t];
  float inv = 1.0f / (c > 1.0f ? c : 1.0f);
  float h1[64];
  for (int j = 0; j < 64; ++j) {
    float s = fb1[j];
    for (int k = 0; k < 128; ++k)
      s += gsum[t * 128 + k] * inv * fW1[k * 64 + j];
    h1[j] = s > 0.0f ? s : 0.0f;
  }
  float o0 = fb2[0], o1 = fb2[1];
  for (int j = 0; j < 64; ++j) {
    o0 += h1[j] * fW2[j * 2 + 0];
    o1 += h1[j] * fW2[j * 2 + 1];
  }
  float m = o0 > o1 ? o0 : o1;
  float e0 = __expf(o0 - m), e1 = __expf(o1 - m);
  float s = e0 + e1;
  out[t * 2 + 0] = e0 / s;
  out[t * 2 + 1] = e1 / s;
}

// ---------------------------------------------------------------------------
extern "C" void kernel_launch(void* const* d_in, const int* in_sizes, int n_in,
                              void* d_out, int out_size, void* d_ws, size_t ws_size,
                              hipStream_t stream) {
  const int N = in_sizes[0];
  const int E = in_sizes[1] / 2;
  const int G = out_size / 2;

  const int*   x_idx = (const int*)d_in[0];
  const int*   ei    = (const int*)d_in[1];      // [0..E) = src, [E..2E) = dst
  const float* eattr = (const float*)d_in[2];
  const int*   batch = (const int*)d_in[3];
  const float* emb   = (const float*)d_in[4];

  const float *eps[3], *W1[3], *B1[3], *W2[3], *B2[3];
  for (int l = 0; l < 3; ++l) {
    int base = 5 + l * 5;
    eps[l] = (const float*)d_in[base + 0];
    W1[l]  = (const float*)d_in[base + 1];
    B1[l]  = (const float*)d_in[base + 2];
    W2[l]  = (const float*)d_in[base + 3];
    B2[l]  = (const float*)d_in[base + 4];
  }
  const float* fW1 = (const float*)d_in[20];
  const float* fb1 = (const float*)d_in[21];
  const float* fW2 = (const float*)d_in[22];
  const float* fb2 = (const float*)d_in[23];

  // workspace layout
  char* ws = (char*)d_ws;
  const size_t XB   = (size_t)N * 128 * sizeof(float);
  const size_t XOFF = 0;
  const size_t HOFF = (XB + 255) & ~(size_t)255;
  size_t WOFF = (HOFF + XB + 255) & ~(size_t)255;
  const size_t WPK  = 4 * 8 * 32 * 16;                       // halves per matrix
  size_t GOFF = (WOFF + 6 * WPK * sizeof(_Float16) + 255) & ~(size_t)255;
  size_t COFF = GOFF + (size_t)G * 128 * sizeof(float);

  float*     X    = (float*)(ws + XOFF);
  float*     H    = (float*)(ws + HOFF);
  _Float16*  Wp   = (_Float16*)(ws + WOFF);
  float*     gsum = (float*)(ws + GOFF);
  float*     cnt  = (float*)(ws + COFF);

  const int totalND = N * 128;
  const long totalED = (long)E * 128;

  // 1. embedding gather
  gin_embed<<<(totalND + 255) / 256, 256, 0, stream>>>(x_idx, emb, X, totalND);

  // 2. pack all six 128x128 weight matrices into WMMA B layout
  for (int l = 0; l < 3; ++l) {
    gin_pack_w<<<64, 256, 0, stream>>>(W1[l], Wp + (size_t)(2 * l + 0) * WPK);
    gin_pack_w<<<64, 256, 0, stream>>>(W2[l], Wp + (size_t)(2 * l + 1) * WPK);
  }

  // 3. GIN layers
  const int mlpBlocks = (N + 15) / 16;
  for (int l = 0; l < 3; ++l) {
    gin_init_h<<<(totalND + 255) / 256, 256, 0, stream>>>(H, X, eps[l], totalND);
    gin_scatter<<<(int)((totalED + 255) / 256), 256, 0, stream>>>(
        H, X, ei, ei + E, eattr, totalED);
    gin_mlp<<<mlpBlocks, 256, 0, stream>>>(
        H, Wp + (size_t)(2 * l + 0) * WPK, B1[l],
        Wp + (size_t)(2 * l + 1) * WPK, B2[l], X, N);
  }

  // 4. pooling + head
  gin_zero_pool<<<(G * 128 + 255) / 256, 256, 0, stream>>>(gsum, cnt, G);
  gin_pool<<<(totalND + 255) / 256, 256, 0, stream>>>(X, batch, gsum, cnt, totalND);
  gin_head<<<1, ((G + 63) / 64) * 64, 0, stream>>>(gsum, cnt, fW1, fb1, fW2, fb2,
                                                   (float*)d_out, G);
}
